// GNN_52123723104402
// MI455X (gfx1250) — matrix operand order repaired and checked by reading, not compiled, not run on hardware
//
#include <hip/hip_runtime.h>

typedef __attribute__((ext_vector_type(16))) __bf16 v16bf;
typedef __attribute__((ext_vector_type(8)))  __bf16 v8bf;
typedef __attribute__((ext_vector_type(8)))  float  v8f;

#define IN_F  128
#define HID_F 128
#define OUT_F 64
#define KP    136   // padded K pitch for transposed W in LDS: 272B rows, 16B aligned

__device__ __forceinline__ void atomic_add_f32(float* p, float v) {
    __hip_atomic_fetch_add(p, v, __ATOMIC_RELAXED, __HIP_MEMORY_SCOPE_AGENT);
}

__device__ __forceinline__ v16bf cat16(v8bf lo, v8bf hi) {
    return __builtin_shufflevector(lo, hi, 0, 1, 2, 3, 4, 5, 6, 7,
                                           8, 9, 10, 11, 12, 13, 14, 15);
}

// ---------------- degree / normalization ----------------
__global__ __launch_bounds__(256) void k_deg_init(float* deg, int n) {
    int i = blockIdx.x * 256 + threadIdx.x;
    if (i < n) deg[i] = 1.0f;  // self-loop contributes 1
}
__global__ __launch_bounds__(256) void k_deg_count(const int* __restrict__ dst, float* deg, int e) {
    int i = blockIdx.x * 256 + threadIdx.x;
    if (i < e) atomic_add_f32(&deg[dst[i]], 1.0f);
}
__global__ __launch_bounds__(256) void k_dinv(float* deg, int n) {
    int i = blockIdx.x * 256 + threadIdx.x;
    if (i < n) deg[i] = rsqrtf(deg[i]);  // deg >= 1 always
}

// ---------------- layer 1 GEMM: H1 = X @ W1, AGG1 = dinv^2 * H1 ----------------
__global__ __launch_bounds__(256)
void k_gemm1(const float* __restrict__ x, const float* __restrict__ W1,
             const float* __restrict__ dinv, float* __restrict__ H1,
             float* __restrict__ AGG1, int n) {
    __shared__ __bf16 sWt[HID_F * KP];    // ~34 KB, transposed [n][k], padded pitch
    __shared__ __bf16 sA[16 * IN_F];      // 4 KB, [m][k]
    const int  tid  = threadIdx.x;
    const int  m0   = blockIdx.x * 16;
    const bool full = (m0 + 16 <= n);     // uniform: tile fully in range

    // coalesced global read (row-major), transposed LDS write
    for (int i = tid; i < IN_F * HID_F; i += 256) {
        int k = i >> 7, nn = i & (HID_F - 1);
        sWt[nn * KP + k] = (__bf16)W1[i];
    }
    if (full) {
        for (int i = tid; i < 16 * IN_F; i += 256)
            sA[i] = (__bf16)x[(size_t)m0 * IN_F + i];
    } else {
        for (int i = tid; i < 16 * IN_F; i += 256) {
            int row = m0 + (i >> 7);
            sA[i] = (__bf16)(row < n ? x[(size_t)row * IN_F + (i & (IN_F - 1))] : 0.0f);
        }
    }
    __syncthreads();

    const int lane = tid & 31, wave = tid >> 5;
    const int n0   = wave * 16;          // 8 waves cover 128 output cols
    const int arow = lane & 15;
    const int hi   = (lane & 16) ? 1 : 0;
    const int bn   = n0 + arow;

    v8f acc = {};
#pragma unroll
    for (int kk = 0; kk < IN_F / 32; ++kk) {
        const int kb = kk * 32;
        // A fragment (16-bit 16x32, ISA 7.12.2): K {ak..ak+7, ak+16..ak+23}
        const int ak = kb + hi * 8;
        v8bf alo = *(const v8bf*)&sA[arow * IN_F + ak];
        v8bf ahi = *(const v8bf*)&sA[arow * IN_F + ak + 16];
        // B fragment: lane = column, K {bk..bk+7, bk+16..bk+23} (transposed LDS)
        const int bk = kb + hi * 8;
        v8bf blo = *(const v8bf*)&sWt[bn * KP + bk];
        v8bf bhi = *(const v8bf*)&sWt[bn * KP + bk + 16];

        acc = __builtin_amdgcn_wmma_f32_16x16x32_bf16(false, cat16(alo, ahi),
                                                      false, cat16(blo, bhi),
                                                      (short)0, acc, false, false);
    }

    const int rbase = m0 + hi * 8;
    if (full) {
#pragma unroll
        for (int r = 0; r < 8; ++r) {
            int   row = rbase + r;
            float v   = acc[r];
            float di  = dinv[row];
            H1[(size_t)row * HID_F + bn]   = v;
            AGG1[(size_t)row * HID_F + bn] = v * di * di;  // self-loop term
        }
    } else {
#pragma unroll
        for (int r = 0; r < 8; ++r) {
            int row = rbase + r;
            if (row < n) {
                float v  = acc[r];
                float di = dinv[row];
                H1[(size_t)row * HID_F + bn]   = v;
                AGG1[(size_t)row * HID_F + bn] = v * di * di;
            }
        }
    }
}

// ---------------- layer 1 edge scatter: AGG1[dst] += norm * H1[src] ----------------
__global__ __launch_bounds__(256)
void k_scatter1(const int* __restrict__ src, const int* __restrict__ dst,
                const float* __restrict__ dinv, const float* __restrict__ H1,
                float* __restrict__ AGG1, int e) {
    int eidx = blockIdx.x * 8 + (threadIdx.x >> 5);
    if (eidx >= e) return;
    int lane = threadIdx.x & 31;
    int s = src[eidx], d = dst[eidx];
    float nm = dinv[s] * dinv[d];
    float4 v = ((const float4*)(H1 + (size_t)s * HID_F))[lane];
    float* op = AGG1 + (size_t)d * HID_F + lane * 4;
    atomic_add_f32(op + 0, v.x * nm);
    atomic_add_f32(op + 1, v.y * nm);
    atomic_add_f32(op + 2, v.z * nm);
    atomic_add_f32(op + 3, v.w * nm);
}

// ---------------- layer 2 GEMM: H2 = relu(AGG1+b1) @ W2, out = b2 + dinv^2*H2 ----------------
__global__ __launch_bounds__(128)
void k_gemm2(const float* __restrict__ AGG1, const float* __restrict__ b1,
             const float* __restrict__ W2, const float* __restrict__ b2,
             const float* __restrict__ dinv, float* __restrict__ H2,
             float* __restrict__ out, int n) {
    __shared__ __bf16 sWt[OUT_F * KP];    // ~17 KB, transposed [n][k]
    __shared__ __bf16 sA[16 * HID_F];     // 4 KB
    const int  tid  = threadIdx.x;
    const int  m0   = blockIdx.x * 16;
    const bool full = (m0 + 16 <= n);

    for (int i = tid; i < HID_F * OUT_F; i += 128) {
        int k = i >> 6, nn = i & (OUT_F - 1);
        sWt[nn * KP + k] = (__bf16)W2[i];
    }
    if (full) {
        for (int i = tid; i < 16 * HID_F; i += 128) {
            int c = i & (HID_F - 1);
            sA[i] = (__bf16)fmaxf(AGG1[(size_t)m0 * HID_F + i] + b1[c], 0.0f);
        }
    } else {
        for (int i = tid; i < 16 * HID_F; i += 128) {
            int c = i & (HID_F - 1);
            int row = m0 + (i >> 7);
            float v = (row < n) ? (AGG1[(size_t)row * HID_F + c] + b1[c]) : 0.0f;
            sA[i] = (__bf16)fmaxf(v, 0.0f);   // fused bias + ReLU
        }
    }
    __syncthreads();

    const int lane = tid & 31, wave = tid >> 5;
    const int n0   = wave * 16;          // 4 waves cover 64 output cols
    const int arow = lane & 15;
    const int hi   = (lane & 16) ? 1 : 0;
    const int bn   = n0 + arow;

    v8f acc = {};
#pragma unroll
    for (int kk = 0; kk < HID_F / 32; ++kk) {
        const int kb = kk * 32;
        const int ak = kb + hi * 8;
        v8bf alo = *(const v8bf*)&sA[arow * HID_F + ak];
        v8bf ahi = *(const v8bf*)&sA[arow * HID_F + ak + 16];
        const int bk = kb + hi * 8;
        v8bf blo = *(const v8bf*)&sWt[bn * KP + bk];
        v8bf bhi = *(const v8bf*)&sWt[bn * KP + bk + 16];

        acc = __builtin_amdgcn_wmma_f32_16x16x32_bf16(false, cat16(alo, ahi),
                                                      false, cat16(blo, bhi),
                                                      (short)0, acc, false, false);
    }

    const int rbase = m0 + hi * 8;
    if (full) {
#pragma unroll
        for (int r = 0; r < 8; ++r) {
            int   row = rbase + r;
            float v   = acc[r];
            float di  = dinv[row];
            H2[(size_t)row * OUT_F + bn]  = v;
            out[(size_t)row * OUT_F + bn] = b2[bn] + v * di * di;  // bias + self-loop
        }
    } else {
#pragma unroll
        for (int r = 0; r < 8; ++r) {
            int row = rbase + r;
            if (row < n) {
                float v  = acc[r];
                float di = dinv[row];
                H2[(size_t)row * OUT_F + bn]  = v;
                out[(size_t)row * OUT_F + bn] = b2[bn] + v * di * di;
            }
        }
    }
}

// ---------------- layer 2 edge scatter: out[dst] += norm * H2[src] ----------------
__global__ __launch_bounds__(256)
void k_scatter2(const int* __restrict__ src, const int* __restrict__ dst,
                const float* __restrict__ dinv, const float* __restrict__ H2,
                float* __restrict__ out, int e) {
    int eidx = blockIdx.x * 8 + (threadIdx.x >> 5);
    if (eidx >= e) return;
    int lane = threadIdx.x & 31;
    int s = src[eidx], d = dst[eidx];
    float nm = dinv[s] * dinv[d];
    float2 v = ((const float2*)(H2 + (size_t)s * OUT_F))[lane];
    float* op = out + (size_t)d * OUT_F + lane * 2;
    atomic_add_f32(op + 0, v.x * nm);
    atomic_add_f32(op + 1, v.y * nm);
}

extern "C" void kernel_launch(void* const* d_in, const int* in_sizes, int n_in,
                              void* d_out, int out_size, void* d_ws, size_t ws_size,
                              hipStream_t stream) {
    const float* x  = (const float*)d_in[0];
    const float* W1 = (const float*)d_in[1];
    const float* b1 = (const float*)d_in[2];
    const float* W2 = (const float*)d_in[3];
    const float* b2 = (const float*)d_in[4];
    const int*   ei = (const int*)d_in[5];

    const int n = in_sizes[0] / IN_F;
    const int e = in_sizes[5] / 2;
    const int* src = ei;
    const int* dst = ei + e;

    // Workspace layout (all L2-resident on MI455X: ~128 MB < 192 MB L2)
    float* ws   = (float*)d_ws;
    size_t np   = (size_t)((n + 255) & ~255);
    float* dinv = ws;                      // [n]   (deg -> dinv in place)
    float* H1   = dinv + np;               // [n,128]
    float* AGG1 = H1   + np * HID_F;       // [n,128]
    float* H2   = AGG1 + np * HID_F;       // [n,64]

    const int nb  = (n + 255) / 256;
    const int eb  = (e + 255) / 256;
    const int ewb = (e + 7) / 8;           // wave per edge, 8 waves/block
    const int mb  = (n + 15) / 16;         // 16-row GEMM tiles

    k_deg_init <<<nb,  256, 0, stream>>>(dinv, n);
    k_deg_count<<<eb,  256, 0, stream>>>(dst, dinv, e);
    k_dinv     <<<nb,  256, 0, stream>>>(dinv, n);

    k_gemm1    <<<mb,  256, 0, stream>>>(x, W1, dinv, H1, AGG1, n);
    k_scatter1 <<<ewb, 256, 0, stream>>>(src, dst, dinv, H1, AGG1, e);

    k_gemm2    <<<mb,  128, 0, stream>>>(AGG1, b1, W2, b2, dinv, H2, (float*)d_out, n);
    k_scatter2 <<<ewb, 256, 0, stream>>>(src, dst, dinv, H2, (float*)d_out, e);
}
